// GAT_90666759619101
// MI455X (gfx1250) — matrix operand order repaired and checked by reading, not compiled, not run on hardware
//
#include <hip/hip_runtime.h>
#include <cstdint>
#include <cstddef>

// ---------------------------------------------------------------------------
// GAT forward for MI455X (gfx1250, wave32, WMMA + TDM).
//   h1 = relu(x @ W_e1 + b_e1)          fp32->bf16 on load, bf16 WMMA, bf16 out
//   h2 = h1 @ W_e2 + b_e2               A-tile via tensor_load_to_lds (TDM)
//   h3 = h2 @ W_lin + b_lin             A-tile via TDM, f32 out
//   hh = h3 @ W_att ; a_src/a_dst       tiny, VALU
//   segment softmax + aggregation       edge-parallel, atomics
// ---------------------------------------------------------------------------

typedef __attribute__((ext_vector_type(16))) __bf16 bf16x16;
typedef __attribute__((ext_vector_type(8)))  __bf16 bf16x8;
typedef __attribute__((ext_vector_type(8)))  float  f32x8;
typedef unsigned int u32x4 __attribute__((ext_vector_type(4)));
typedef int          i32x8 __attribute__((ext_vector_type(8)));
typedef int          i32x4 __attribute__((ext_vector_type(4)));

union FragU { bf16x16 v; bf16x8 h[2]; };

#if __has_builtin(__builtin_amdgcn_tensor_load_to_lds)
#define HAVE_TDM 1
#else
#define HAVE_TDM 0
#endif

#if HAVE_TDM
// DMA a BM(128) x BK(32) bf16 tile (row-major, row stride K elements) from
// global memory into LDS at byte offset ldsOff, inserting 16B of padding after
// every 64B row so the LDS row stride is 80B (= LDSK of 40 elements).
// Rows >= mRem read as zero (tensor OOB), matching the zero-fill fallback.
__device__ __forceinline__ void tdm_load_a_tile(const __bf16* gbase,
                                                unsigned ldsOff, int mRem, int K)
{
    const unsigned long long ga = (unsigned long long)(uintptr_t)gbase;
    u32x4 g0;
    g0.x = 1u;                                          // count=1 valid descriptor
    g0.y = ldsOff;                                      // lds_addr (bytes)
    g0.z = (unsigned)(ga & 0xFFFFFFFFull);              // global_addr[31:0]
    g0.w = (unsigned)((ga >> 32) & 0x01FFFFFFull)       // global_addr[56:32]
         | (2u << 30);                                  // type=2 ("image")

    const unsigned mlo = (unsigned)mRem & 0xFFFFu;
    const unsigned mhi = ((unsigned)mRem >> 16) & 0xFFFFu;
    i32x8 g1;
    g1[0] = (int)((1u << 16)        // data_size = 2 bytes
                | (1u << 20)        // pad_enable
                | (3u << 22)        // pad_interval: 16 DWORDs (=64B row)
                | (3u << 25));      // pad_amount:    4 DWORDs (=16B pad)
    g1[1] = (int)(32u << 16);       // tensor_dim0 = 32 (bits 79:48, low half)
    g1[2] = (int)(mlo << 16);       // tensor_dim0 hi=0 | tensor_dim1 lo
    g1[3] = (int)(mhi | (32u << 16)); // tensor_dim1 hi | tile_dim0 = 32
    g1[4] = 128;                    // tile_dim1 = 128 rows, tile_dim2 = 0
    g1[5] = K;                      // tensor_dim0_stride[31:0] (elements)
    g1[6] = 0;                      // stride hi | tensor_dim1_stride lo
    g1[7] = 0;
    const i32x4 z4 = {0, 0, 0, 0};
#if defined(__clang_major__) && (__clang_major__ >= 23)
    const i32x8 z8 = {0, 0, 0, 0, 0, 0, 0, 0};
    __builtin_amdgcn_tensor_load_to_lds(g0, g1, z4, z4, z8, 0);
#else
    __builtin_amdgcn_tensor_load_to_lds(g0, g1, z4, z4, 0);
#endif
}
#endif

// ---------------------------------------------------------------------------
// Tiled GEMM:  C[M,N] = act(A[M,K] @ B[K,N] + bias)
// Block tile: BM=128 x BN, BK=32. 8 waves (WROWS x WCOLS), each wave computes
// TM x TN WMMA 16x16 tiles with v_wmma_f32_16x16x32_bf16.
// ---------------------------------------------------------------------------
template<int BN, int TM, int TN, int WROWS, int WCOLS,
         bool A_BF16, bool OUT_BF16, bool RELU>
__global__ __launch_bounds__(256)
void gemm_wmma_bf16(const void* __restrict__ Ap,
                    const float* __restrict__ Bp,
                    const float* __restrict__ biasp,
                    void* __restrict__ Cp,
                    int M, int K, int N)
{
    constexpr int BM   = 128;
    constexpr int BK   = 32;
    constexpr int LDSK = BK + 8;   // padded element stride vs bank conflicts

    __shared__ __bf16 sA[BM * LDSK];   // row-major [m][k], 80B row stride
    __shared__ __bf16 sB[BN * LDSK];   // column-major [n][k], 80B col stride

    const int tid   = threadIdx.x;
    const int lane  = tid & 31;
    const int wave  = tid >> 5;
    const int wrow  = wave % WROWS;
    const int wcol  = wave / WROWS;
    const int mBase = blockIdx.y * BM;
    const int nBase = blockIdx.x * BN;
    const int waveM = wrow * TM * 16;
    const int waveN = wcol * TN * 16;
    const int half  = lane >> 4;       // 0: lanes 0-15, 1: lanes 16-31
    const int l15   = lane & 15;

#if HAVE_TDM
    const unsigned sAoff = (unsigned)(unsigned long long)(uintptr_t)&sA[0];
#endif

    f32x8 acc[TM][TN] = {};

    const int kTiles = K / BK;
    for (int kt = 0; kt < kTiles; ++kt) {
        // ---- A tile (BM x 32) -> sA -------------------------------------
        if constexpr (A_BF16) {
#if HAVE_TDM
            if (wave == 0) {
                tdm_load_a_tile((const __bf16*)Ap + (size_t)mBase * K + kt * BK,
                                sAoff, M - mBase, K);
            }
#else
            const int r  = tid >> 1;
            const int c0 = (tid & 1) * 16;
            const int gr = mBase + r;
            __bf16* dp = &sA[r * LDSK + c0];
            if (gr < M) {
                const __bf16* sp = (const __bf16*)Ap + (size_t)gr * K + kt * BK + c0;
                uint4 v0 = *(const uint4*)(sp);
                uint4 v1 = *(const uint4*)(sp + 8);
                *(uint4*)(dp)     = v0;
                *(uint4*)(dp + 8) = v1;
            } else {
                uint4 z = {0u, 0u, 0u, 0u};
                *(uint4*)(dp)     = z;
                *(uint4*)(dp + 8) = z;
            }
#endif
        } else {
            const int r  = tid >> 1;
            const int c0 = (tid & 1) * 16;
            const int gr = mBase + r;
            __bf16* dp = &sA[r * LDSK + c0];
            if (gr < M) {
                const float* sp = (const float*)Ap + (size_t)gr * K + kt * BK + c0;
                #pragma unroll
                for (int j = 0; j < 16; j += 4) {
                    float4 v = *(const float4*)(sp + j);
                    dp[j + 0] = (__bf16)v.x;
                    dp[j + 1] = (__bf16)v.y;
                    dp[j + 2] = (__bf16)v.z;
                    dp[j + 3] = (__bf16)v.w;
                }
            } else {
                #pragma unroll
                for (int j = 0; j < 16; ++j) dp[j] = (__bf16)0.0f;
            }
        }

        // ---- B tile (32 x BN) -> sB transposed, packed k-runs -----------
        {
            constexpr int KH = (BN >= 128) ? 4 : 2;   // k-values per thread
            constexpr int CH = BN / (8 * KH);         // columns per thread
            constexpr int KG = BK / KH;               // thread groups along k
            const int k0 = (tid % KG) * KH;
            const int n0 = (tid / KG) * CH;
            float v[KH][CH];
            #pragma unroll
            for (int kk = 0; kk < KH; ++kk) {
                const float* sp = Bp + (size_t)(kt * BK + k0 + kk) * N + nBase + n0;
                #pragma unroll
                for (int j = 0; j < CH; ++j) v[kk][j] = sp[j];
            }
            #pragma unroll
            for (int j = 0; j < CH; ++j) {
                if constexpr (KH == 4) {
                    union { __bf16 h[4]; uint2 u; } p;
                    p.h[0] = (__bf16)v[0][j]; p.h[1] = (__bf16)v[1][j];
                    p.h[2] = (__bf16)v[2][j]; p.h[3] = (__bf16)v[3][j];
                    *(uint2*)&sB[(n0 + j) * LDSK + k0] = p.u;
                } else {
                    union { __bf16 h[2]; unsigned u; } p;
                    p.h[0] = (__bf16)v[0][j]; p.h[1] = (__bf16)v[1][j];
                    *(unsigned*)&sB[(n0 + j) * LDSK + k0] = p.u;
                }
            }
        }

#if HAVE_TDM
        if constexpr (A_BF16) {
            if (wave == 0) __builtin_amdgcn_s_wait_tensorcnt(0);
        }
#endif
        __syncthreads();

        // ---- build fragments per documented wave32 operand layouts ------
        bf16x16 afrag[TM];
        bf16x16 bfrag[TN];
        #pragma unroll
        for (int i = 0; i < TM; ++i) {
            const int m = waveM + i * 16 + l15;
            FragU u;
            // lanes 0-15: K 0-7 then 16-23 ; lanes 16-31: K 8-15 then 24-31
            u.h[0] = *(const bf16x8*)&sA[m * LDSK + half * 8];
            u.h[1] = *(const bf16x8*)&sA[m * LDSK + 16 + half * 8];
            afrag[i] = u.v;
        }
        #pragma unroll
        for (int j = 0; j < TN; ++j) {
            const int n = waveN + j * 16 + l15;
            FragU u;
            // lanes 0-15: K 0-7 then 16-23 ; lanes 16-31: K 8-15 then 24-31
            u.h[0] = *(const bf16x8*)&sB[n * LDSK + half * 8];
            u.h[1] = *(const bf16x8*)&sB[n * LDSK + 16 + half * 8];
            bfrag[j] = u.v;
        }

        #pragma unroll
        for (int i = 0; i < TM; ++i)
            #pragma unroll
            for (int j = 0; j < TN; ++j)
                acc[i][j] = __builtin_amdgcn_wmma_f32_16x16x32_bf16(
                    false, afrag[i], false, bfrag[j],
                    (short)0, acc[i][j], false, false);

        __syncthreads();
    }

    // ---- epilogue: bias (+ relu), store ----
    #pragma unroll
    for (int i = 0; i < TM; ++i) {
        #pragma unroll
        for (int j = 0; j < TN; ++j) {
            const int n = nBase + waveN + j * 16 + l15;
            const float bv = biasp[n];
            #pragma unroll
            for (int r = 0; r < 8; ++r) {
                const int m = mBase + waveM + i * 16 + half * 8 + r;
                if (m < M) {
                    float v = acc[i][j][r] + bv;
                    if (RELU) v = fmaxf(v, 0.0f);
                    if (OUT_BF16)
                        ((__bf16*)Cp)[(size_t)m * N + n] = (__bf16)v;
                    else
                        ((float*)Cp)[(size_t)m * N + n] = v;
                }
            }
        }
    }
}

// ---------------------------------------------------------------------------
// Node prep: hh = h3 @ W_att  (32 -> 6), a_src/a_dst attention coefficients
// ---------------------------------------------------------------------------
__global__ __launch_bounds__(256)
void node_prep(const float* __restrict__ h3, const float* __restrict__ Watt,
               const float* __restrict__ attS, const float* __restrict__ attD,
               float* __restrict__ hh, float* __restrict__ aS,
               float* __restrict__ aD, int N)
{
    __shared__ float sW[32 * 6];
    __shared__ float sS[6];
    __shared__ float sD[6];
    const int tid = threadIdx.x;
    if (tid < 192) sW[tid] = Watt[tid];
    if (tid < 6) { sS[tid] = attS[tid]; sD[tid] = attD[tid]; }
    __syncthreads();

    const int i = blockIdx.x * 256 + tid;
    if (i >= N) return;

    float hv[32];
    const float4* hp = (const float4*)(h3 + (size_t)i * 32);
    #pragma unroll
    for (int q = 0; q < 8; ++q) {
        float4 v = hp[q];
        hv[q * 4 + 0] = v.x; hv[q * 4 + 1] = v.y;
        hv[q * 4 + 2] = v.z; hv[q * 4 + 3] = v.w;
    }
    float o[6] = {0.f, 0.f, 0.f, 0.f, 0.f, 0.f};
    #pragma unroll
    for (int k = 0; k < 32; ++k) {
        #pragma unroll
        for (int c = 0; c < 6; ++c) o[c] += hv[k] * sW[k * 6 + c];
    }
    float s0 = 0.f, s1 = 0.f, d0 = 0.f, d1 = 0.f;
    #pragma unroll
    for (int c = 0; c < 3; ++c) {
        s0 += o[c] * sS[c];     s1 += o[3 + c] * sS[3 + c];
        d0 += o[c] * sD[c];     d1 += o[3 + c] * sD[3 + c];
    }
    #pragma unroll
    for (int c = 0; c < 6; ++c) hh[(size_t)i * 6 + c] = o[c];
    aS[i * 2] = s0; aS[i * 2 + 1] = s1;
    aD[i * 2] = d0; aD[i * 2 + 1] = d1;
}

// ---------------------------------------------------------------------------
// Edge-parallel segment softmax + aggregation
// ---------------------------------------------------------------------------
__device__ __forceinline__ float lrelu(float x) { return x > 0.f ? x : 0.2f * x; }

// order-preserving float->uint map so atomicMax(uint) == float max
__device__ __forceinline__ unsigned enc_f32(float f) {
    unsigned b = __float_as_uint(f);
    return b ^ ((b & 0x80000000u) ? 0xFFFFFFFFu : 0x80000000u);
}
__device__ __forceinline__ float dec_f32(unsigned u) {
    unsigned b = u ^ ((u & 0x80000000u) ? 0x80000000u : 0xFFFFFFFFu);
    return __uint_as_float(b);
}

__global__ void init_buffers(float* denom, unsigned* amaxu, float* accum, int N)
{
    const int i = blockIdx.x * 256 + threadIdx.x;
    if (i < N * 6) accum[i] = 0.f;
    if (i < N * 2) { denom[i] = 0.f; amaxu[i] = 0u; }
}

__global__ void attn_max(const int* __restrict__ src, const int* __restrict__ dst,
                         const float* __restrict__ aS, const float* __restrict__ aD,
                         unsigned* __restrict__ amaxu, int E, int ETOT)
{
    const int e = blockIdx.x * 256 + threadIdx.x;
    if (e >= ETOT) return;
    int s, d;
    if (e < E) { s = src[e]; d = dst[e]; } else { s = d = e - E; }
    const float a0 = lrelu(aS[s * 2]     + aD[d * 2]);
    const float a1 = lrelu(aS[s * 2 + 1] + aD[d * 2 + 1]);
    atomicMax(&amaxu[d * 2],     enc_f32(a0));
    atomicMax(&amaxu[d * 2 + 1], enc_f32(a1));
}

__global__ void decode_max(const unsigned* __restrict__ u, float* __restrict__ f, int n)
{
    const int i = blockIdx.x * 256 + threadIdx.x;
    if (i < n) f[i] = dec_f32(u[i]);
}

__global__ void attn_sum(const int* __restrict__ src, const int* __restrict__ dst,
                         const float* __restrict__ aS, const float* __restrict__ aD,
                         const float* __restrict__ amaxf, float* __restrict__ denom,
                         int E, int ETOT)
{
    const int e = blockIdx.x * 256 + threadIdx.x;
    if (e >= ETOT) return;
    int s, d;
    if (e < E) { s = src[e]; d = dst[e]; } else { s = d = e - E; }
    const float a0 = lrelu(aS[s * 2]     + aD[d * 2]);
    const float a1 = lrelu(aS[s * 2 + 1] + aD[d * 2 + 1]);
    atomicAdd(&denom[d * 2],     expf(a0 - amaxf[d * 2]));
    atomicAdd(&denom[d * 2 + 1], expf(a1 - amaxf[d * 2 + 1]));
}

__global__ void attn_agg(const int* __restrict__ src, const int* __restrict__ dst,
                         const float* __restrict__ aS, const float* __restrict__ aD,
                         const float* __restrict__ amaxf, const float* __restrict__ denom,
                         const float* __restrict__ hh, float* __restrict__ accum,
                         int E, int ETOT)
{
    const int e = blockIdx.x * 256 + threadIdx.x;
    if (e >= ETOT) return;
    int s, d;
    if (e < E) { s = src[e]; d = dst[e]; } else { s = d = e - E; }
    const float a0 = lrelu(aS[s * 2]     + aD[d * 2]);
    const float a1 = lrelu(aS[s * 2 + 1] + aD[d * 2 + 1]);
    const float w0 = expf(a0 - amaxf[d * 2])     / denom[d * 2];
    const float w1 = expf(a1 - amaxf[d * 2 + 1]) / denom[d * 2 + 1];
    const float* hs = hh + (size_t)s * 6;
    atomicAdd(&accum[d * 6 + 0], w0 * hs[0]);
    atomicAdd(&accum[d * 6 + 1], w0 * hs[1]);
    atomicAdd(&accum[d * 6 + 2], w0 * hs[2]);
    atomicAdd(&accum[d * 6 + 3], w1 * hs[3]);
    atomicAdd(&accum[d * 6 + 4], w1 * hs[4]);
    atomicAdd(&accum[d * 6 + 5], w1 * hs[5]);
}

__global__ void finalize(const float* __restrict__ accum,
                         const float* __restrict__ bias,
                         float* __restrict__ out, int N)
{
    const int i = blockIdx.x * 256 + threadIdx.x;
    if (i >= N) return;
    const float* a = accum + (size_t)i * 6;
    out[i * 3 + 0] = 0.5f * (a[0] + a[3]) + bias[0];
    out[i * 3 + 1] = 0.5f * (a[1] + a[4]) + bias[1];
    out[i * 3 + 2] = 0.5f * (a[2] + a[5]) + bias[2];
}

// ---------------------------------------------------------------------------
extern "C" void kernel_launch(void* const* d_in, const int* in_sizes, int n_in,
                              void* d_out, int out_size, void* d_ws, size_t ws_size,
                              hipStream_t stream)
{
    const float* x    = (const float*)d_in[0];
    const int*   ei   = (const int*)  d_in[1];
    const float* W1   = (const float*)d_in[2];
    const float* b1   = (const float*)d_in[3];
    const float* W2   = (const float*)d_in[4];
    const float* b2   = (const float*)d_in[5];
    const float* Wl   = (const float*)d_in[6];
    const float* bl   = (const float*)d_in[7];
    const float* Wa   = (const float*)d_in[8];
    const float* attS = (const float*)d_in[9];
    const float* attD = (const float*)d_in[10];
    const float* bias = (const float*)d_in[11];
    float* out = (float*)d_out;

    const int N    = in_sizes[0] / 768;
    const int E    = in_sizes[1] / 2;
    const int ETOT = E + N;

    // workspace carve-out (~170 MB total)
    uint8_t* ws = (uint8_t*)d_ws;
    size_t off = 0;
    auto take = [&](size_t bytes) -> uint8_t* {
        uint8_t* p = ws + off;
        off = (off + bytes + 255) & ~(size_t)255;
        return p;
    };
    __bf16*   h1  = (__bf16*)  take((size_t)N * 512 * 2);
    __bf16*   h2  = (__bf16*)  take((size_t)N * 256 * 2);
    float*    h3  = (float*)   take((size_t)N * 32 * 4);
    float*    hh  = (float*)   take((size_t)N * 6 * 4);
    float*    aS  = (float*)   take((size_t)N * 2 * 4);
    float*    aD  = (float*)   take((size_t)N * 2 * 4);
    unsigned* amu = (unsigned*)take((size_t)N * 2 * 4);
    float*    amf = (float*)   take((size_t)N * 2 * 4);
    float*    den = (float*)   take((size_t)N * 2 * 4);
    float*    acc = (float*)   take((size_t)N * 6 * 4);
    (void)ws_size; (void)n_in; (void)out_size;

    const dim3 blk(256);
    const int  gm = (N + 127) / 128;

    // h1 = relu(x @ W_e1 + b_e1)   [N,768] x [768,512]
    gemm_wmma_bf16<128, 2, 4, 4, 2, false, true,  true >
        <<<dim3(512 / 128, gm), blk, 0, stream>>>(x,  W1, b1, h1, N, 768, 512);
    // h2 = h1 @ W_e2 + b_e2        [N,512] x [512,256]
    gemm_wmma_bf16<128, 2, 4, 4, 2, true,  true,  false>
        <<<dim3(256 / 128, gm), blk, 0, stream>>>(h1, W2, b2, h2, N, 512, 256);
    // h3 = h2 @ W_lin + b_lin      [N,256] x [256,32]
    gemm_wmma_bf16<32,  1, 2, 8, 1, true,  false, false>
        <<<dim3(1, gm),         blk, 0, stream>>>(h2, Wl, bl, h3, N, 256, 32);

    node_prep<<<(N + 255) / 256, blk, 0, stream>>>(h3, Wa, attS, attD, hh, aS, aD, N);
    init_buffers<<<(N * 6 + 255) / 256, blk, 0, stream>>>(den, amu, acc, N);

    const int ge = (ETOT + 255) / 256;
    attn_max <<<ge, blk, 0, stream>>>(ei, ei + E, aS, aD, amu, E, ETOT);
    decode_max<<<(N * 2 + 255) / 256, blk, 0, stream>>>(amu, amf, N * 2);
    attn_sum <<<ge, blk, 0, stream>>>(ei, ei + E, aS, aD, amf, den, E, ETOT);
    attn_agg <<<ge, blk, 0, stream>>>(ei, ei + E, aS, aD, amf, den, hh, acc, E, ETOT);
    finalize <<<(N + 255) / 256, blk, 0, stream>>>(acc, bias, out, N);
}
